// _STANAttention_44787918962892
// MI455X (gfx1250) — compile-verified, matrix operand order checked by reading, not run on hardware
//
#include <hip/hip_runtime.h>
#include <hip/hip_bf16.h>

// Problem constants (B,S,D,H from reference)
enum { Bn = 8, Sn = 1024, Dn = 1024, Hn = 16, HDn = 64 };

typedef __attribute__((ext_vector_type(16))) __bf16 bf16x16;
typedef __attribute__((ext_vector_type(8)))  __bf16 bf16x8;
typedef __attribute__((ext_vector_type(8)))  float  f32x8;
typedef __attribute__((ext_vector_type(4)))  float  f32x4;
typedef __attribute__((ext_vector_type(4)))  int    i32x4;

// address-space qualified pointee types for the async-copy builtin
typedef __attribute__((address_space(1))) i32x4 as1_i32x4;
typedef __attribute__((address_space(3))) i32x4 as3_i32x4;

union ABfrag { bf16x16 v; bf16x8 h[2]; };

// ---------------------------------------------------------------- async copy
// GLOBAL_LOAD_ASYNC_TO_LDS_B128 (ASYNCcnt-tracked) when the builtin exists;
// otherwise fall back to a register bounce (global_load_b128 + ds_store_b128).
#if __has_builtin(__builtin_amdgcn_global_load_async_to_lds_b128)
#define HAVE_ASYNC_LDS 1
#else
#define HAVE_ASYNC_LDS 0
#endif

__device__ __forceinline__ void copy16_to_lds(const unsigned short* g, unsigned short* l) {
#if HAVE_ASYNC_LDS
    __builtin_amdgcn_global_load_async_to_lds_b128(
        (as1_i32x4*)g, (as3_i32x4*)l, 0, 0);
#else
    *reinterpret_cast<f32x4*>(l) = *reinterpret_cast<const f32x4*>(g);
#endif
}

__device__ __forceinline__ void copies_wait() {
#if HAVE_ASYNC_LDS
#if __has_builtin(__builtin_amdgcn_s_wait_asynccnt)
    __builtin_amdgcn_s_wait_asynccnt(0);
#endif
#endif
}

// ---------------------------------------------------------------- helpers
__device__ __forceinline__ unsigned short f2bfbits(float f) {
    union { float f; unsigned u; } a; a.f = f;
    unsigned u = a.u;
    unsigned r = u + 0x7FFFu + ((u >> 16) & 1u);   // round-to-nearest-even
    return (unsigned short)(r >> 16);
}

__device__ __forceinline__ f32x8 zero8() {
    f32x8 z;
#pragma unroll
    for (int i = 0; i < 8; ++i) z[i] = 0.0f;
    return z;
}

// Load one 16x32 bf16 WMMA A-fragment (or 32x16 B-fragment, symmetric layout)
// from a row-major [16 rows x ld] bf16 (ushort) source (global or LDS).
// Per ISA 7.12.2: lane L<16 -> row L, K = {kb..kb+7, kb+16..kb+23};
//                 lane L>=16 -> row L-16, K base kb+8.
__device__ __forceinline__ bf16x16 load_frag(const unsigned short* base, int ld, int kbase) {
    const int lane = threadIdx.x & 31;
    const int r    = lane & 15;
    const int ko   = kbase + ((lane >> 4) << 3);
    const unsigned short* p = base + r * ld + ko;
    ABfrag f;
    f.h[0] = *reinterpret_cast<const bf16x8*>(p);
    f.h[1] = *reinterpret_cast<const bf16x8*>(p + 16);
    return f.v;
}

__device__ __forceinline__ f32x8 wmma_bf16(bf16x16 a, bf16x16 b, f32x8 c) {
    return __builtin_amdgcn_wmma_f32_16x16x32_bf16(false, a, false, b, (short)0, c, false, false);
}

// ---------------------------------------------------------------- converts
__global__ void __launch_bounds__(256) cvt_kernel(const float* __restrict__ src,
                                                  unsigned short* __restrict__ dst, int n) {
    int i = blockIdx.x * 256 + threadIdx.x;
    if (i < n) dst[i] = f2bfbits(src[i]);
}

// ---------------------------------------------------------------- GEMM core
// out[m,n] = sum_k A[m,k] * W[n,k]; M tile 128 (8 waves x 16), N tile 64.
// B (weight) tile 64x32 staged in LDS, double-buffered, async-copied.
// Each of the 256 threads copies one 16-byte chunk per K step.
template <typename EPI>
__device__ __forceinline__ void gemm_128x64(
    const unsigned short* __restrict__ arows,  // [.., K] row-major, rows m0..m0+127
    const unsigned short* __restrict__ wrows,  // [.., K] row-major, rows n0..n0+63
    int K, EPI epi)
{
    __shared__ unsigned short bls[2][64 * 32];

    const int tid  = threadIdx.x;
    const int wave = tid >> 5;
    const int crow = tid >> 2;           // 0..63   (copy row)
    const int ckof = (tid & 3) * 8;      // 0,8,16,24

    f32x8 acc[4];
#pragma unroll
    for (int f = 0; f < 4; ++f) acc[f] = zero8();

    const unsigned short* arow = arows + (size_t)(wave * 16) * K;
    const unsigned short* wsrc = wrows + (size_t)crow * K + ckof;

    // prologue: stage k=0
    copy16_to_lds(wsrc, &bls[0][crow * 32 + ckof]);

    for (int k = 0; k < K; k += 32) {
        const int st = (k >> 5) & 1;
        copies_wait();
        __syncthreads();                                  // stage `st` ready
        if (k + 32 < K)
            copy16_to_lds(wsrc + k + 32, &bls[st ^ 1][crow * 32 + ckof]);

        // batch all fragment loads (one clause, one wait), then 4 WMMAs
        bf16x16 a = load_frag(arow, K, k);
        bf16x16 bfr[4];
#pragma unroll
        for (int f = 0; f < 4; ++f)
            bfr[f] = load_frag(&bls[st][f * 16 * 32], 32, 0);
#pragma unroll
        for (int f = 0; f < 4; ++f)
            acc[f] = wmma_bf16(a, bfr[f], acc[f]);
    }
    epi(acc);
}

// ---------------------------------------------------------------- QKV GEMM
// M=8192, N=3072, K=1024. Epilogue scatters to Q (scaled), K, V^T (bf16).
__global__ void __launch_bounds__(256) qkv_gemm_kernel(
    const unsigned short* __restrict__ xb,   // [8192,1024] bf16
    const unsigned short* __restrict__ wb,   // [3072,1024] bf16
    const float* __restrict__ bias,          // [3072]
    unsigned short* __restrict__ qb,         // [B,H,S,HD]
    unsigned short* __restrict__ kbuf,       // [B,H,S,HD]
    unsigned short* __restrict__ vtb)        // [B,H,HD,S]
{
    const int lane = threadIdx.x & 31;
    const int wave = threadIdx.x >> 5;
    const int col  = lane & 15;
    const int half = lane >> 4;
    const int n0   = blockIdx.x * 64;
    const int m0   = blockIdx.y * 128;

    // 128-row M tile never crosses a batch boundary (1024 % 128 == 0)
    const int b  = m0 / Sn;
    const int s0 = (m0 % Sn) + wave * 16 + 8 * half;

    const int t  = n0 / Dn;          // 0=q, 1=k, 2=v
    const int h  = (n0 % Dn) / HDn;

    gemm_128x64(xb + (size_t)m0 * Dn, wb + (size_t)n0 * Dn, Dn,
        [&](f32x8 (&acc)[4]) {
#pragma unroll
            for (int f = 0; f < 4; ++f) {
                const int hd  = f * 16 + col;
                const float bia = bias[n0 + f * 16 + col];
#pragma unroll
                for (int r = 0; r < 8; ++r) {
                    const int s = s0 + r;
                    float v = acc[f][r] + bia;
                    if (t == 0)
                        qb[((size_t)(b * Hn + h) * Sn + s) * HDn + hd] = f2bfbits(v * 0.125f);
                    else if (t == 1)
                        kbuf[((size_t)(b * Hn + h) * Sn + s) * HDn + hd] = f2bfbits(v);
                    else
                        vtb[((size_t)(b * Hn + h) * HDn + hd) * Sn + s] = f2bfbits(v);
                }
            }
        });
}

// ---------------------------------------------------------------- out proj
// M=8192, N=1024, K=1024, fp32 output + bias.
__global__ void __launch_bounds__(256) oproj_gemm_kernel(
    const unsigned short* __restrict__ ab,   // [8192,1024] bf16
    const unsigned short* __restrict__ wb,   // [1024,1024] bf16
    const float* __restrict__ bias,          // [1024]
    float* __restrict__ out)                 // [8192,1024] f32
{
    const int lane = threadIdx.x & 31;
    const int wave = threadIdx.x >> 5;
    const int col  = lane & 15;
    const int half = lane >> 4;
    const int n0   = blockIdx.x * 64;
    const int m0   = blockIdx.y * 128 + wave * 16 + 8 * half;

    gemm_128x64(ab + (size_t)(blockIdx.y * 128) * Dn, wb + (size_t)n0 * Dn, Dn,
        [&](f32x8 (&acc)[4]) {
#pragma unroll
            for (int f = 0; f < 4; ++f) {
                const int n = n0 + f * 16 + col;
                const float bia = bias[n];
#pragma unroll
                for (int r = 0; r < 8; ++r)
                    out[(size_t)(m0 + r) * Dn + n] = acc[f][r] + bia;
            }
        });
}

// ---------------------------------------------------------------- attention
// Flash attention: one wave per 16-row Q tile; 8 waves/block cover 128 rows of
// one (b,h). All waves share async-staged K (32x64) and V^T (64x32) LDS tiles,
// double-buffered. Per 32-key chunk: QK^T via 2x(2 WMMA, K=64), online softmax
// (shfl_xor over 16-lane groups), P re-fragmented via per-wave LDS, PV via
// 4 WMMA (K=32, N=64).
__global__ void __launch_bounds__(256) attn_kernel(
    const unsigned short* __restrict__ qb,     // [B,H,S,HD] bf16 (pre-scaled)
    const unsigned short* __restrict__ kbuf,   // [B,H,S,HD] bf16
    const unsigned short* __restrict__ vtb,    // [B,H,HD,S] bf16
    const float* __restrict__ pair_bias,       // [H,S,S] f32
    const int* __restrict__ pmask,             // [B,S]
    unsigned short* __restrict__ aout)         // [B,S,D] bf16
{
    __shared__ unsigned short kls[2][32 * 64];    // [key][hd]
    __shared__ unsigned short vls[2][64 * 32];    // [hd][key]
    __shared__ unsigned short plds[8][16 * 32];   // per-wave P tile

    const int tid  = threadIdx.x;
    const int wave = tid >> 5;
    const int lane = tid & 31;
    const int col  = lane & 15;
    const int half = lane >> 4;

    const int tilesPerBH = Sn / 128;              // 8
    const int bh = blockIdx.x / tilesPerBH;
    const int it = blockIdx.x % tilesPerBH;
    const int b  = bh / Hn, h = bh % Hn;
    const int i0 = it * 128 + wave * 16;

    const unsigned short* qrow   = qb   + ((size_t)(b * Hn + h) * Sn + i0) * HDn;
    const unsigned short* kbase  = kbuf + ((size_t)(b * Hn + h) * Sn) * HDn;
    const unsigned short* vtbase = vtb  + ((size_t)(b * Hn + h) * HDn) * Sn;

    // copy-lane assignments (one b128 per thread per tile)
    const int krow = tid >> 3, kkof = (tid & 7) * 8;   // K tile: 32 rows x 64
    const int vrow = tid >> 2, vjof = (tid & 3) * 8;   // V tile: 64 rows x 32

    const bf16x16 qa0 = load_frag(qrow, HDn, 0);
    const bf16x16 qa1 = load_frag(qrow, HDn, 32);

    f32x8 oacc[4];
#pragma unroll
    for (int f = 0; f < 4; ++f) oacc[f] = zero8();
    float mrow[8], lrow[8];
#pragma unroll
    for (int r = 0; r < 8; ++r) { mrow[r] = -3.0e38f; lrow[r] = 0.0f; }

    unsigned short* myp = plds[wave];

    // prologue: stage j0 = 0
    copy16_to_lds(kbase + (size_t)krow * HDn + kkof, &kls[0][krow * 64 + kkof]);
    copy16_to_lds(vtbase + (size_t)vrow * Sn + vjof, &vls[0][vrow * 32 + vjof]);

    for (int j0 = 0; j0 < Sn; j0 += 32) {
        const int st = (j0 >> 5) & 1;
        copies_wait();
        __syncthreads();                              // stage `st` ready
        if (j0 + 32 < Sn) {
            copy16_to_lds(kbase + (size_t)(j0 + 32 + krow) * HDn + kkof,
                          &kls[st ^ 1][krow * 64 + kkof]);
            copy16_to_lds(vtbase + (size_t)vrow * Sn + (j0 + 32) + vjof,
                          &vls[st ^ 1][vrow * 32 + vjof]);
        }

        int mk[2];
        mk[0] = pmask[b * Sn + j0 + col];
        mk[1] = pmask[b * Sn + j0 + 16 + col];

        // batch all 4 K-fragment loads, then the 4 QK^T WMMAs
        bf16x16 kfr[4];
#pragma unroll
        for (int sub = 0; sub < 2; ++sub) {
            kfr[sub * 2 + 0] = load_frag(&kls[st][(sub * 16) * 64], 64, 0);
            kfr[sub * 2 + 1] = load_frag(&kls[st][(sub * 16) * 64], 64, 32);
        }

        f32x8 sc[2];
#pragma unroll
        for (int sub = 0; sub < 2; ++sub) {
            f32x8 c = zero8();
            c = wmma_bf16(qa0, kfr[sub * 2 + 0], c);
            c = wmma_bf16(qa1, kfr[sub * 2 + 1], c);
            const float* pb = pair_bias + ((size_t)h * Sn + (i0 + 8 * half)) * Sn
                              + j0 + sub * 16 + col;
#pragma unroll
            for (int r = 0; r < 8; ++r) {
                float v = c[r] + pb[(size_t)r * Sn];
                sc[sub][r] = mk[sub] ? -3.0e38f : v;
            }
        }

        // online softmax (rows in vgpr slot r + 8*half; cols span 16 lanes)
#pragma unroll
        for (int r = 0; r < 8; ++r) {
            float mx = fmaxf(sc[0][r], sc[1][r]);
#pragma unroll
            for (int o = 1; o < 16; o <<= 1) mx = fmaxf(mx, __shfl_xor(mx, o, 32));
            const float mnew  = fmaxf(mrow[r], mx);
            const float alpha = __expf(mrow[r] - mnew);
            float p0 = mk[0] ? 0.0f : __expf(sc[0][r] - mnew);
            float p1 = mk[1] ? 0.0f : __expf(sc[1][r] - mnew);
            sc[0][r] = p0; sc[1][r] = p1;
            float ls = p0 + p1;
#pragma unroll
            for (int o = 1; o < 16; o <<= 1) ls += __shfl_xor(ls, o, 32);
            lrow[r] = lrow[r] * alpha + ls;
            mrow[r] = mnew;
#pragma unroll
            for (int f = 0; f < 4; ++f) oacc[f][r] *= alpha;
        }

        // P (C layout) -> per-wave LDS 16x32 bf16 -> A fragment (same wave)
#pragma unroll
        for (int sub = 0; sub < 2; ++sub)
#pragma unroll
            for (int r = 0; r < 8; ++r)
                myp[(r + 8 * half) * 32 + sub * 16 + col] = f2bfbits(sc[sub][r]);

        // batch P + all 4 V-fragment loads, then the 4 PV WMMAs
        const bf16x16 pa = load_frag(myp, 32, 0);
        bf16x16 vfr[4];
#pragma unroll
        for (int f = 0; f < 4; ++f)
            vfr[f] = load_frag(&vls[st][(f * 16) * 32], 32, 0);
#pragma unroll
        for (int f = 0; f < 4; ++f)
            oacc[f] = wmma_bf16(pa, vfr[f], oacc[f]);
    }

    // epilogue: normalize and store to [B,S,D]
#pragma unroll
    for (int r = 0; r < 8; ++r) {
        const float inv = 1.0f / lrow[r];
        const int m = i0 + r + 8 * half;
        unsigned short* orow = aout + ((size_t)b * Sn + m) * Dn + h * HDn;
#pragma unroll
        for (int f = 0; f < 4; ++f)
            orow[f * 16 + col] = f2bfbits(oacc[f][r] * inv);
    }
}

// ---------------------------------------------------------------- launch
extern "C" void kernel_launch(void* const* d_in, const int* in_sizes, int n_in,
                              void* d_out, int out_size, void* d_ws, size_t ws_size,
                              hipStream_t stream) {
    const float* x         = (const float*)d_in[0];   // [8,1024,1024]
    const float* qkv_w     = (const float*)d_in[1];   // [3072,1024]
    const float* qkv_b     = (const float*)d_in[2];   // [3072]
    const float* out_w     = (const float*)d_in[3];   // [1024,1024]
    const float* out_b     = (const float*)d_in[4];   // [1024]
    const float* pair_bias = (const float*)d_in[5];   // [16,1024,1024]
    const int*   pmask     = (const int*)d_in[6];     // [8,1024]

    char* ws = (char*)d_ws;
    size_t off = 0;
    auto carve = [&](size_t bytes) {
        void* p = ws + off;
        off += (bytes + 255) & ~(size_t)255;
        return p;
    };
    const size_t nX   = (size_t)Bn * Sn * Dn;        // 8,388,608
    const size_t nWQ  = (size_t)3 * Dn * Dn;         // 3,145,728
    const size_t nWO  = (size_t)Dn * Dn;             // 1,048,576
    const size_t nQKV = (size_t)Bn * Hn * Sn * HDn;  // 8,388,608

    unsigned short* xb  = (unsigned short*)carve(2 * nX);
    unsigned short* wqb = (unsigned short*)carve(2 * nWQ);
    unsigned short* wob = (unsigned short*)carve(2 * nWO);
    unsigned short* qb  = (unsigned short*)carve(2 * nQKV);
    unsigned short* kb  = (unsigned short*)carve(2 * nQKV);
    unsigned short* vtb = (unsigned short*)carve(2 * nQKV);
    unsigned short* aob = (unsigned short*)carve(2 * nX);

    cvt_kernel<<<(int)((nX  + 255) / 256), 256, 0, stream>>>(x,     xb,  (int)nX);
    cvt_kernel<<<(int)((nWQ + 255) / 256), 256, 0, stream>>>(qkv_w, wqb, (int)nWQ);
    cvt_kernel<<<(int)((nWO + 255) / 256), 256, 0, stream>>>(out_w, wob, (int)nWO);

    qkv_gemm_kernel<<<dim3(3 * Dn / 64, (Bn * Sn) / 128), 256, 0, stream>>>(
        xb, wqb, qkv_b, qb, kb, vtb);

    attn_kernel<<<Bn * Hn * (Sn / 128), 256, 0, stream>>>(
        qb, kb, vtb, pair_bias, pmask, aob);

    oproj_gemm_kernel<<<dim3(Dn / 64, (Bn * Sn) / 128), 256, 0, stream>>>(
        aob, wob, out_b, (float*)d_out);
}
